// ArchitecturallyCorrectRNN_90486370993052
// MI455X (gfx1250) — matrix operand order repaired and checked by reading, not compiled, not run on hardware
//
#include <hip/hip_runtime.h>
#include <hip/hip_bf16.h>

// ---- problem constants (match reference) ----
#define N_IN   8192
#define STATE_ 512
#define OUT_N  64
#define BATCH_ 256
#define KBAND  81                  // k = int(N*N*0.01)//N = 81
#define BANDW  128                 // 4 x K=32 chunks; window cols g = i0..i0+127
#define N_ROWS (N_IN + OUT_N)
#define WA_NNZ (N_IN * KBAND)      // 663552 structured band nonzeros

typedef __attribute__((ext_vector_type(16))) _Float16 v16h;
typedef __attribute__((ext_vector_type(8)))  float    v8f;

// ------------------------------------------------------------------
// Kernel 1: zero the z accumulator [N_ROWS x BATCH] in workspace
// ------------------------------------------------------------------
__global__ void zero_ws(float* __restrict__ z, int n) {
  int i = blockIdx.x * blockDim.x + threadIdx.x;
  if (i < n) z[i] = 0.0f;
}

// ------------------------------------------------------------------
// Kernel 2: random-COO remainder (Ws / Ma / Ms blocks, ~7% of nnz)
// 64 threads per edge, 4 batch elements per thread, f32 atomics.
// ------------------------------------------------------------------
__global__ void coo_scatter(const float* __restrict__ a_t,
                            const float* __restrict__ s_t,
                            const int*   __restrict__ rows,
                            const int*   __restrict__ cols,
                            const float* __restrict__ vals,
                            float* __restrict__ z, int nrest) {
  int t  = blockIdx.x * blockDim.x + threadIdx.x;
  int e  = t >> 6;                // edge index within remainder
  int bq = (t & 63) << 2;         // batch group of 4
  if (e >= nrest) return;
  e += WA_NNZ;
  const int   r = rows[e];
  const int   c = cols[e];
  const float v = vals[e];
  const float* xcol;
  int stride;
  if (c < N_IN) { xcol = a_t + c;          stride = N_IN;  }
  else          { xcol = s_t + (c - N_IN); stride = STATE_; }
  float* zr = z + (size_t)r * BATCH_ + bq;
#pragma unroll
  for (int i = 0; i < 4; ++i)
    atomicAdd(zr + i, v * xcol[(size_t)(bq + i) * stride]);
}

// ------------------------------------------------------------------
// Kernel 3: structured band block as dense-band GEMM on the WMMA pipe.
// Block = 16 output rows x 256 batch. 8 waves, each owns TWO 16x16
// batch tiles (two accumulators -> independent back-to-back WMMAs).
// The uniform wrap test is hoisted AROUND two fully-unrolled loops so
// the 505/512 common blocks get a branch-free straight-line path of
// ds_load_b128 / global_load_b128 / v_cvt_pk_f16_f32 / 8x v_wmma.
// ------------------------------------------------------------------
__global__ __launch_bounds__(256)
void band_wmma(const float* __restrict__ a_t,
               const float* __restrict__ values,
               const float* __restrict__ z,
               float* __restrict__ out) {
  __shared__ _Float16 Ah[16 * BANDW];          // 4 KB: band tile, f16
  __shared__ float    tileLds[16][16][17];     // 17 KB: transpose, pitch 17

  const int i0 = blockIdx.x << 4;              // first row of this strip

  // densify band weights: Ah[m][c] = w(i0+m, i0+c) for c-m in [1..81], else 0
  for (int idx = threadIdx.x; idx < 16 * BANDW; idx += 256) {
    int m   = idx >> 7;                        // / BANDW
    int c   = idx & (BANDW - 1);
    int off = c - m;                           // j in [1,81] inside the band
    float v = 0.0f;
    if (off >= 1 && off <= KBAND)
      v = values[(size_t)(i0 + m) * KBAND + (off - 1)];
    Ah[idx] = (_Float16)v;
  }
  __syncthreads();

  const int lane = threadIdx.x & 31;
  const int wave = threadIdx.x >> 5;           // 0..7
  const int h    = lane >> 4;                  // lane half (K-split)
  const int ln   = lane & 15;                  // A-row / B-column within tile
  const int b0   = wave << 5;                  // this wave's 32-batch slab

  const float* arow0 = a_t + (size_t)(b0 + ln) * N_IN;   // tile 0 batch row
  const float* arow1 = arow0 + (size_t)16 * N_IN;        // tile 1 batch row

  v8f acc0 = {}, acc1 = {};

  if (i0 + BANDW <= N_IN) {
    // ---------- hot path (no mod-N wrap): fully unrolled, vector loads ----
#pragma unroll
    for (int kk = 0; kk < BANDW; kk += 32) {
      // A fragment: contiguous k order, same as B (half h: K 16h+0..15)
      v16h af;
#pragma unroll
      for (int e = 0; e < 16; ++e) {
        int k = 16 * h + e;
        af[e] = Ah[ln * BANDW + kk + k];
      }
      // B fragments: lanes 0-15 hold K=0..15, lanes 16-31 K=16..31; N = ln
      v16h bf0, bf1;
      const int gstart = i0 + kk + 16 * h;     // multiple of 16 -> b128 aligned
#pragma unroll
      for (int q = 0; q < 4; ++q) {
        float4 x0 = *(const float4*)(arow0 + gstart + 4 * q);
        float4 x1 = *(const float4*)(arow1 + gstart + 4 * q);
        bf0[4*q+0] = (_Float16)x0.x; bf0[4*q+1] = (_Float16)x0.y;
        bf0[4*q+2] = (_Float16)x0.z; bf0[4*q+3] = (_Float16)x0.w;
        bf1[4*q+0] = (_Float16)x1.x; bf1[4*q+1] = (_Float16)x1.y;
        bf1[4*q+2] = (_Float16)x1.z; bf1[4*q+3] = (_Float16)x1.w;
      }
      acc0 = __builtin_amdgcn_wmma_f32_16x16x32_f16(
          false, af, false, bf0, (short)0, acc0, false, false);
      acc1 = __builtin_amdgcn_wmma_f32_16x16x32_f16(
          false, af, false, bf1, (short)0, acc1, false, false);
    }
  } else {
    // ---------- rare path (last 7 blocks): scalar gathers with wrap -------
#pragma unroll
    for (int kk = 0; kk < BANDW; kk += 32) {
      v16h af;
#pragma unroll
      for (int e = 0; e < 16; ++e) {
        int k = 16 * h + e;
        af[e] = Ah[ln * BANDW + kk + k];
      }
      v16h bf0, bf1;
      const int gstart = i0 + kk + 16 * h;
#pragma unroll
      for (int e = 0; e < 16; ++e) {
        int g = gstart + e;
        g -= (g >= N_IN) ? N_IN : 0;           // (i+j) mod N wraparound
        bf0[e] = (_Float16)arow0[g];
        bf1[e] = (_Float16)arow1[g];
      }
      acc0 = __builtin_amdgcn_wmma_f32_16x16x32_f16(
          false, af, false, bf0, (short)0, acc0, false, false);
      acc1 = __builtin_amdgcn_wmma_f32_16x16x32_f16(
          false, af, false, bf1, (short)0, acc1, false, false);
    }
  }

  // C/D layout: VGPR r -> M = r + 8h, N = ln.  Add COO partial, sigmoid.
#pragma unroll
  for (int r = 0; r < 8; ++r) {
    int row = i0 + r + 8 * h;
    const float* zrow = z + (size_t)row * BATCH_ + b0 + ln;
    float z0 = zrow[0]  + acc0[r];
    float z1 = zrow[16] + acc1[r];
    tileLds[wave * 2 + 0][r + 8 * h][ln] = 1.0f / (1.0f + __expf(-z0));
    tileLds[wave * 2 + 1][r + 8 * h][ln] = 1.0f / (1.0f + __expf(-z1));
  }
  __syncthreads();

  // transposed, 64B-coalesced stores: out[b*N + i0 + m]
#pragma unroll
  for (int t = 0; t < 2; ++t) {
#pragma unroll
    for (int u = 0; u < 8; ++u) {
      int bl = u * 2 + h;                      // batch-local 0..15
      out[(size_t)(b0 + 16 * t + bl) * N_IN + i0 + ln] =
          tileLds[wave * 2 + t][ln][bl];
    }
  }
}

// ------------------------------------------------------------------
// Kernel 4: the 64 "M" rows (COO-only): 2*sigmoid(z)-1
// ------------------------------------------------------------------
__global__ void m_activation(const float* __restrict__ z, float* __restrict__ out) {
  int i = blockIdx.x * blockDim.x + threadIdx.x;
  if (i >= OUT_N * BATCH_) return;
  int b = i / OUT_N, o = i % OUT_N;
  float zz = z[(size_t)(N_IN + o) * BATCH_ + b];
  float s  = 1.0f / (1.0f + __expf(-zz));
  out[(size_t)BATCH_ * N_IN + i] = 2.0f * s - 1.0f;
}

// ------------------------------------------------------------------
extern "C" void kernel_launch(void* const* d_in, const int* in_sizes, int n_in,
                              void* d_out, int out_size, void* d_ws, size_t ws_size,
                              hipStream_t stream) {
  const float* a_t  = (const float*)d_in[0];
  const float* s_t  = (const float*)d_in[1];
  const int*   rows = (const int*)  d_in[2];
  const int*   cols = (const int*)  d_in[3];
  const float* vals = (const float*)d_in[4];
  float* out = (float*)d_out;
  float* z   = (float*)d_ws;                   // [N_ROWS x BATCH] f32 = 8.45 MB

  const int nnz   = in_sizes[2];
  const int nrest = nnz - WA_NNZ;

  const int zn = N_ROWS * BATCH_;
  zero_ws<<<(zn + 255) / 256, 256, 0, stream>>>(z, zn);

  if (nrest > 0) {
    long long tw = (long long)nrest * 64;      // 64 threads per edge
    coo_scatter<<<(unsigned)((tw + 255) / 256), 256, 0, stream>>>(
        a_t, s_t, rows, cols, vals, z, nrest);
  }

  band_wmma<<<N_IN / 16, 256, 0, stream>>>(a_t, vals, z, out);

  m_activation<<<(OUT_N * BATCH_ + 255) / 256, 256, 0, stream>>>(z, out);
}
